// Encoder_25048249270385
// MI455X (gfx1250) — matrix operand order repaired and checked by reading, not compiled, not run on hardware
//
#include <hip/hip_runtime.h>

// ---------------------------------------------------------------------------
// 2-layer GCN + FC for MI455X (gfx1250, wave32).
//
// Memory-bound op: the edge scatter (800k edges) dominates. We exploit
// linearity S(XW) = (SX)W to aggregate layer-1 BEFORE its GEMM so both
// scatters move only 128 floats/edge (~1.6 GB random traffic @ 23.3 TB/s).
// GEMMs (8.2 GFLOP total) run on v_wmma_f32_16x16x32_bf16 — bf16 inputs,
// f32 accumulation — one wave per 16x16 output tile.
// ---------------------------------------------------------------------------

typedef __attribute__((ext_vector_type(16))) __bf16 v16bf;
typedef __attribute__((ext_vector_type(8)))  __bf16 v8bf;
typedef __attribute__((ext_vector_type(8)))  float  v8f;

__device__ __forceinline__ unsigned short f2bf(float f) {
    unsigned u = __float_as_uint(f);
    u += 0x7fffu + ((u >> 16) & 1u);          // round-to-nearest-even
    return (unsigned short)(u >> 16);
}

// ---------------- degree / normalization ----------------
__global__ void deg_init_k(float* deg, int n) {
    int i = blockIdx.x * blockDim.x + threadIdx.x;
    if (i < n) deg[i] = 1.0f;                 // self-loop contribution
}
__global__ void deg_scatter_k(const long long* __restrict__ dst, float* deg, int E) {
    int e = blockIdx.x * blockDim.x + threadIdx.x;
    if (e < E) atomicAdd(&deg[(int)dst[e]], 1.0f);
}
__global__ void deg_rsqrt_k(float* deg, int n) {
    int i = blockIdx.x * blockDim.x + threadIdx.x;
    if (i < n) deg[i] = rsqrtf(deg[i]);       // deg >= 1 always
}

// ---------------- S * H  (F = 128) ----------------
// self term: out[i,f] = h[i,f] * dinv[i]^2
__global__ void self_term_k(const float* __restrict__ h, const float* __restrict__ dinv,
                            float* __restrict__ out, int total) {
    int idx = blockIdx.x * blockDim.x + threadIdx.x;
    if (idx >= total) return;
    float di = dinv[idx >> 7];
    out[idx] = h[idx] * di * di;
}
// one wave per edge; lane handles 4 contiguous floats (float4 load + atomics)
__global__ void edge_scatter128_k(const float* __restrict__ h,
                                  const long long* __restrict__ src,
                                  const long long* __restrict__ dst,
                                  const float* __restrict__ dinv,
                                  float* __restrict__ out, int E) {
    int t    = blockIdx.x * blockDim.x + threadIdx.x;
    int e    = t >> 5;
    int lane = t & 31;
    if (e >= E) return;
    int s = (int)src[e], d = (int)dst[e];
    float norm = dinv[s] * dinv[d];
    float4 v = *((const float4*)(h + (size_t)s * 128) + lane);
    float* o = out + (size_t)d * 128 + lane * 4;
    atomicAdd(o + 0, v.x * norm);
    atomicAdd(o + 1, v.y * norm);
    atomicAdd(o + 2, v.z * norm);
    atomicAdd(o + 3, v.w * norm);
}

// ---------------- conversions ----------------
__global__ void cvt_bf16_k(const float* __restrict__ in, unsigned short* __restrict__ out, int total) {
    int idx = blockIdx.x * blockDim.x + threadIdx.x;
    if (idx < total) out[idx] = f2bf(in[idx]);
}
__global__ void bias_cvt_bf16_k(const float* __restrict__ in, const float* __restrict__ bias,
                                unsigned short* __restrict__ out, int total) {
    int idx = blockIdx.x * blockDim.x + threadIdx.x;
    if (idx < total) out[idx] = f2bf(in[idx] + bias[idx & 127]);
}
// W [K x N] f32  ->  Wt [N x K] bf16  (so B fragments are K-contiguous)
__global__ void w_cvt_t_k(const float* __restrict__ W, unsigned short* __restrict__ Wt,
                          int K, int N) {
    int idx = blockIdx.x * blockDim.x + threadIdx.x;
    if (idx >= K * N) return;
    int k = idx / N, nn = idx - k * N;
    Wt[(size_t)nn * K + k] = f2bf(W[idx]);
}

// ---------------- WMMA GEMM: C[M,N] = A[M,K] * Bt[N,K]^T (+bias)(ReLU) -------
// One wave per 16x16 C tile. A fragment (16x32 bf16): lane = row, halves hold
// K 0-15 / 16-31 -> one contiguous 32B load.
// B fragment (32x16 bf16): lane = column, halves hold K 0-15 / 16-31 -> one
// contiguous 32B load from the transposed weights.
template <int K, bool RELU, bool OUT_BF16>
__global__ void gemm_wmma_k(const __bf16* __restrict__ A,
                            const __bf16* __restrict__ Bt,
                            const float*  __restrict__ bias,
                            void* __restrict__ Cout,
                            int M, int N) {
    int wave = (blockIdx.x * blockDim.x + threadIdx.x) >> 5;
    int lane = threadIdx.x & 31;
    int ntiles = N >> 4;
    int ttiles = (M >> 4) * ntiles;
    if (wave >= ttiles) return;
    int tm = wave / ntiles, tn = wave - tm * ntiles;
    int r = lane & 15, half = lane >> 4;

    const __bf16* Arow = A  + (size_t)(tm * 16 + r) * K;
    const __bf16* Brow = Bt + (size_t)(tn * 16 + r) * K;

    v8f acc = {};
#pragma unroll
    for (int kk = 0; kk < K; kk += 32) {
        v16bf a = *(const v16bf*)(Arow + kk + half * 16);
        v16bf b = *(const v16bf*)(Brow + kk + half * 16);
        acc = __builtin_amdgcn_wmma_f32_16x16x32_bf16(
            /*neg_a=*/false, a, /*neg_b=*/false, b,
            /*c_mod=*/(short)0, acc, /*reuse_a=*/false, /*reuse_b=*/false);
    }

    int col = tn * 16 + r;
    float bv = bias ? bias[col] : 0.0f;
#pragma unroll
    for (int v = 0; v < 8; ++v) {
        int row = tm * 16 + v + half * 8;
        float val = acc[v] + bv;
        if (RELU) val = fmaxf(val, 0.0f);
        if (OUT_BF16)
            ((unsigned short*)Cout)[(size_t)row * N + col] = f2bf(val);
        else
            ((float*)Cout)[(size_t)row * N + col] = val;
    }
}

// ---------------------------------------------------------------------------
static inline size_t align_up(size_t v, size_t a) { return (v + a - 1) & ~(a - 1); }

extern "C" void kernel_launch(void* const* d_in, const int* in_sizes, int n_in,
                              void* d_out, int out_size, void* d_ws, size_t ws_size,
                              hipStream_t stream) {
    const float*     x   = (const float*)d_in[0];
    const long long* ei  = (const long long*)d_in[1];   // int64 [2, E]
    const float*     W1  = (const float*)d_in[2];
    const float*     b1  = (const float*)d_in[3];
    const float*     W2  = (const float*)d_in[4];
    const float*     b2  = (const float*)d_in[5];
    const float*     Wfc = (const float*)d_in[6];
    const float*     bfc = (const float*)d_in[7];

    const int n = in_sizes[0] / 128;     // 50000
    const int E = in_sizes[1] / 2;       // 800000
    const long long* src = ei;
    const long long* dst = ei + E;

    // ---- workspace layout ----
    char* w = (char*)d_ws;
    size_t off = 0;
    float* dinv = (float*)(w + off);            off = align_up(off + (size_t)n * 4, 256);
    float* aggx = (float*)(w + off);            off = align_up(off + (size_t)n * 128 * 4, 256);
    unsigned short* xb   = (unsigned short*)(w + off); off = align_up(off + (size_t)n * 128 * 2, 256);
    unsigned short* W1t  = (unsigned short*)(w + off); off = align_up(off + (size_t)128 * 256 * 2, 256);
    unsigned short* W2t  = (unsigned short*)(w + off); off = align_up(off + (size_t)256 * 128 * 2, 256);
    unsigned short* Wfct = (unsigned short*)(w + off); off = align_up(off + (size_t)128 * 128 * 2, 256);
    unsigned short* a1b  = (unsigned short*)(w + off); off = align_up(off + (size_t)n * 256 * 2, 256);
    float* h2   = (float*)(w + off);            off = align_up(off + (size_t)n * 128 * 4, 256);
    float* agg2 = (float*)(w + off);            off = align_up(off + (size_t)n * 128 * 4, 256);
    unsigned short* a2b = (unsigned short*)(w + off); off = align_up(off + (size_t)n * 128 * 2, 256);

    const int T = 256;
    const int nf  = n * 128;

    // 1) symmetric normalization coefficients
    deg_init_k   <<<(n + T - 1) / T, T, 0, stream>>>(dinv, n);
    deg_scatter_k<<<(E + T - 1) / T, T, 0, stream>>>(dst, dinv, E);
    deg_rsqrt_k  <<<(n + T - 1) / T, T, 0, stream>>>(dinv, n);

    // 2) aggx = S * x   (layer-1 aggregation moved before GEMM: S(XW)=(SX)W)
    self_term_k     <<<(nf + T - 1) / T, T, 0, stream>>>(x, dinv, aggx, nf);
    edge_scatter128_k<<<(E * 32 + T - 1) / T, T, 0, stream>>>(x, src, dst, dinv, aggx, E);

    // 3) bf16 conversions
    cvt_bf16_k<<<(nf + T - 1) / T, T, 0, stream>>>(aggx, xb, nf);
    w_cvt_t_k<<<(128 * 256 + T - 1) / T, T, 0, stream>>>(W1, W1t, 128, 256);
    w_cvt_t_k<<<(256 * 128 + T - 1) / T, T, 0, stream>>>(W2, W2t, 256, 128);
    w_cvt_t_k<<<(128 * 128 + T - 1) / T, T, 0, stream>>>(Wfc, Wfct, 128, 128);

    // 4) GEMM1: a1b = relu(aggx @ W1 + b1)   [n x 256] bf16 out
    {
        int tiles = (n / 16) * (256 / 16);
        gemm_wmma_k<128, true, true><<<(tiles + 7) / 8, T, 0, stream>>>(
            (const __bf16*)xb, (const __bf16*)W1t, b1, a1b, n, 256);
    }
    // 5) GEMM2: h2 = a1b @ W2               [n x 128] f32 out (bias after agg)
    {
        int tiles = (n / 16) * (128 / 16);
        gemm_wmma_k<256, false, false><<<(tiles + 7) / 8, T, 0, stream>>>(
            (const __bf16*)a1b, (const __bf16*)W2t, (const float*)nullptr, h2, n, 128);
    }
    // 6) agg2 = S * h2
    self_term_k      <<<(nf + T - 1) / T, T, 0, stream>>>(h2, dinv, agg2, nf);
    edge_scatter128_k<<<(E * 32 + T - 1) / T, T, 0, stream>>>(h2, src, dst, dinv, agg2, E);

    // 7) a2b = bf16(agg2 + b2)
    bias_cvt_bf16_k<<<(nf + T - 1) / T, T, 0, stream>>>(agg2, b2, a2b, nf);

    // 8) GEMM3: out = a2b @ Wfc + bfc       [n x 128] f32 -> d_out
    {
        int tiles = (n / 16) * (128 / 16);
        gemm_wmma_k<128, false, false><<<(tiles + 7) / 8, T, 0, stream>>>(
            (const __bf16*)a2b, (const __bf16*)Wfct, bfc, d_out, n, 128);
    }

    (void)n_in; (void)out_size; (void)ws_size;
}
